// Decoder_17179869184418
// MI455X (gfx1250) — matrix operand order repaired and checked by reading, not compiled, not run on hardware
//
#include <hip/hip_runtime.h>
#include <hip/hip_bf16.h>
#include <math.h>

typedef _Float16 f16;
typedef _Float16 v16h __attribute__((ext_vector_type(16)));
typedef _Float16 v8h  __attribute__((ext_vector_type(8)));
typedef float    v8f  __attribute__((ext_vector_type(8)));
typedef float    v4f  __attribute__((ext_vector_type(4)));

#define NQ       262144      // 64^3 queries per batch
#define DGRID    96
#define DSTRIDE  884736      // 96^3 (channel stride in context_v)
#define CCH      32
#define OUTC     45
#define HID      62          // INTERNAL
#define WAVES    8
#define SH       72          // LDS row stride in f16 (144B, 16B-aligned)

// workspace layout in f16 elements (fragment-ordered weights)
#define MLP_OFF  0           // 6 matrices * 8 frags * 512
#define RES_OFF  24576       // 2 matrices * 4 frags * 512
#define POST_OFF 28672       // 3 frags * 512
#define WF_TOTAL 30208

__device__ __forceinline__ v8f wmma16(v16h a, v16h b, v8f c) {
  return __builtin_amdgcn_wmma_f32_16x16x32_f16(false, a, false, b, (short)0, c,
                                                false, false);
}

// Build a 16x32 f16 A-fragment from the LDS activation tile.
// A 16x32 layout (matches B fragments): lanes 0-15 hold M=0..15, K=0..15;
// lanes 16-31 hold M=0..15, K=16..31.
__device__ __forceinline__ v16h loadA(const f16* hb, int kt, int lh, int ll) {
  int base = ll * SH + kt * 32 + lh * 16;
  v8h c0 = *(const v8h*)(hb + base);        // K+0..7  (16B aligned)
  v8h c1 = *(const v8h*)(hb + base + 8);    // K+8..15
  v16h a;
#pragma unroll
  for (int t = 0; t < 8; t++) { a[t] = c0[t]; a[t + 8] = c1[t]; }
  return a;
}

__device__ __forceinline__ float silu(float x) {
  return x * __builtin_amdgcn_rcpf(1.f + __expf(-x));
}

// ---------------------------------------------------------------------------
// Prep: convert f32 weights to f16, padded, in this kernel's B-fragment order.
// B-fragment (32x16, K x N): lane = N (lane%16); lanes 0-15 hold K 0..15,
// lanes 16-31 hold K 16..31; each lane's 16 K-values contiguous (32B/lane).
// ---------------------------------------------------------------------------
__global__ void prep_weights(const float* __restrict__ mlpW,   // (2,3,62,62)
                             const float* __restrict__ resW,   // (2,62,32)
                             const float* __restrict__ postW,  // (32,45)
                             f16* __restrict__ wf) {
  int t = blockIdx.x * blockDim.x + threadIdx.x;
  if (t >= WF_TOTAL) return;
  float v = 0.f;
  if (t < RES_OFF) {
    int mat = t / 4096;           // blk*3 + layer
    int r = t % 4096;
    int frag = r / 512;           // kt*4 + nt
    int q = r % 512;
    int lane = q / 16, i = q % 16;
    int kt = frag >> 2, nt = frag & 3;
    int k = kt * 32 + (lane >> 4) * 16 + i;
    int n = nt * 16 + (lane & 15);
    if (k < HID && n < HID) v = mlpW[(mat * HID + k) * HID + n];
  } else if (t < POST_OFF) {
    int r2 = t - RES_OFF;
    int mat = r2 / 2048;          // blk
    int r = r2 % 2048;
    int frag = r / 512;           // kt*2 + nt
    int q = r % 512;
    int lane = q / 16, i = q % 16;
    int kt = frag >> 1, nt = frag & 1;
    int k = kt * 32 + (lane >> 4) * 16 + i;
    int n = nt * 16 + (lane & 15);
    if (k < HID && n < CCH) v = resW[(mat * HID + k) * CCH + n];
  } else {
    int r3 = t - POST_OFF;
    int nt = r3 / 512;
    int q = r3 % 512;
    int lane = q / 16, i = q % 16;
    int k = (lane >> 4) * 16 + i;
    int n = nt * 16 + (lane & 15);
    if (k < CCH && n < OUTC) v = postW[k * OUTC + n];
  }
  wf[t] = (f16)v;
}

// ---------------------------------------------------------------------------
// Main decoder: one wave = 16 query rows; 8 waves / block.
// ---------------------------------------------------------------------------
__global__ __launch_bounds__(256) void decoder_main(
    const float* __restrict__ ctx,     // (2,32,96,96,96)
    const float* __restrict__ query,   // (2,NQ,3)
    const float* __restrict__ Amat,    // (2,4,4)
    const float* __restrict__ voxsz,   // (2,3)
    const float* __restrict__ mlpB,    // (2,3,62)
    const float* __restrict__ resB,    // (2,32)
    const float* __restrict__ postB,   // (45)
    const f16* __restrict__ wf,
    float* __restrict__ out)           // (2,45,NQ)
{
  __shared__ __align__(16) f16   hbuf_s[WAVES][16 * SH];
  __shared__ __align__(16) f16   cbuf_s[WAVES][16 * 32];
  __shared__ float               wbuf_s[WAVES][16];
  __shared__ int                 ibuf_s[WAVES][16][4];

  const int wid = threadIdx.x >> 5;
  const int lane = threadIdx.x & 31;
  const int lh = lane >> 4;            // lane half (0/1)
  const int ll = lane & 15;            // lane-in-half

  f16* hb = hbuf_s[wid];
  f16* cb = cbuf_s[wid];
  float* wb = wbuf_s[wid];
  int (*ib)[4] = ibuf_s[wid];

  const int g = blockIdx.x * WAVES + wid;   // tile id, exact cover
  const int rowbase = g * 16;
  const int b = rowbase / NQ;
  const int qbase = rowbase % NQ;

  // small hoisted biases (7 regs total)
  float bias_res[2][2], bias_post[3];
#pragma unroll
  for (int blk = 0; blk < 2; blk++)
#pragma unroll
    for (int nt = 0; nt < 2; nt++)
      bias_res[blk][nt] = resB[blk * CCH + nt * 16 + ll];
#pragma unroll
  for (int nt = 0; nt < 3; nt++) {
    int col = nt * 16 + ll;
    bias_post[nt] = (col < OUTC) ? postB[col] : 0.f;
  }

  // -------- per-row setup (lanes 0..15, one row each) --------
  float qwx = 0, qwy = 0, qwz = 0;           // query world coord
  float sbx = 0, sby = 0, sbz = 0;           // q_sub (frac)
  float bwx = 0, bwy = 0, bwz = 0;           // bot_world
  float vsx = 0, vsy = 0, vsz = 0;           // vox size
  if (lane < 16) {
    int qi = qbase + lane;
    const float* qp = query + ((size_t)b * NQ + qi) * 3;
    qwx = qp[0]; qwy = qp[1]; qwz = qp[2];
    const float* A = Amat + b * 16;
    float a00 = A[0], a01 = A[1], a02 = A[2],  t0 = A[3];
    float a10 = A[4], a11 = A[5], a12 = A[6],  t1 = A[7];
    float a20 = A[8], a21 = A[9], a22 = A[10], t2 = A[11];
    float det = a00 * (a11 * a22 - a12 * a21)
              - a01 * (a10 * a22 - a12 * a20)
              + a02 * (a10 * a21 - a11 * a20);
    float id = 1.0f / det;
    float i00 =  (a11 * a22 - a12 * a21) * id;
    float i01 = -(a01 * a22 - a02 * a21) * id;
    float i02 =  (a01 * a12 - a02 * a11) * id;
    float i10 = -(a10 * a22 - a12 * a20) * id;
    float i11 =  (a00 * a22 - a02 * a20) * id;
    float i12 = -(a00 * a12 - a02 * a10) * id;
    float i20 =  (a10 * a21 - a11 * a20) * id;
    float i21 = -(a00 * a21 - a01 * a20) * id;
    float i22 =  (a00 * a11 - a01 * a10) * id;
    float dx = qwx - t0, dy = qwy - t1, dz = qwz - t2;
    float vx = i00 * dx + i01 * dy + i02 * dz;
    float vy = i10 * dx + i11 * dy + i12 * dz;
    float vz = i20 * dx + i21 * dy + i22 * dz;
    float fx = floorf(vx), fy = floorf(vy), fz = floorf(vz);
    sbx = vx - fx; sby = vy - fy; sbz = vz - fz;
    bwx = a00 * fx + a01 * fy + a02 * fz + t0;
    bwy = a10 * fx + a11 * fy + a12 * fz + t1;
    bwz = a20 * fx + a21 * fy + a22 * fz + t2;
    vsx = voxsz[b * 3 + 0]; vsy = voxsz[b * 3 + 1]; vsz = voxsz[b * 3 + 2];
    ib[lane][0] = (int)fx; ib[lane][1] = (int)fy; ib[lane][2] = (int)fz;
  }
  __syncthreads();

  // hoisted gather base: this lane owns row ll, channels {2*cg + lh}
  size_t gbase;
  {
    int ix0 = ib[ll][0], iy0 = ib[ll][1], iz0 = ib[ll][2];
    gbase = (size_t)b * CCH * DSTRIDE +
            ((size_t)ix0 * DGRID + iy0) * DGRID + iz0 +
            (size_t)lh * DSTRIDE;
  }

  v8f outacc[3];
#pragma unroll
  for (int nt = 0; nt < 3; nt++)
#pragma unroll
    for (int vi = 0; vi < 8; vi++) outacc[nt][vi] = 0.f;

  v8f yacc[2];
  const float FREQ[4] = {6.2831853f, 8.2691431f, 10.882796f, 14.322576f}; // 2π·3^(m/4)

#pragma unroll 1
  for (int corner = 0; corner < 8; corner++) {
    const int oi = (corner >> 2) & 1, oj = (corner >> 1) & 1, ok = corner & 1;
    const int coff = (oi * DGRID + oj) * DGRID + ok;   // corner offset in elems

    // coordinate features + trilinear weight (lanes 0..15)
    if (lane < 16) {
      f16* crow = cb + lane * 32;
      crow[0] = (f16)(bwx + oi * vsx);
      crow[1] = (f16)(bwy + oj * vsy);
      crow[2] = (f16)(bwz + ok * vsz);
      crow[3] = (f16)qwx; crow[4] = (f16)qwy; crow[5] = (f16)qwz;
      float rel[3] = { (sbx - oi + 1.f) * 0.5f,
                       (sby - oj + 1.f) * 0.5f,
                       (sbz - ok + 1.f) * 0.5f };
#pragma unroll
      for (int a = 0; a < 3; a++) {
#pragma unroll
        for (int m = 0; m < 4; m++) {
          float ang = rel[a] * FREQ[m];
          crow[6 + a * 8 + m]     = (f16)__cosf(ang);
          crow[6 + a * 8 + 4 + m] = (f16)__sinf(ang);
        }
      }
      crow[30] = (f16)0.f; crow[31] = (f16)0.f;
      float wx = oi ? sbx : 1.f - sbx;
      float wy = oj ? sby : 1.f - sby;
      float wz = ok ? sbz : 1.f - sbz;
      wb[lane] = wx * wy * wz;
    }
    // gather context vector: lane -> row ll, channels 2*cg+lh
    {
      const float* p = ctx + gbase + coff;
#pragma unroll
      for (int cg = 0; cg < 16; cg++)
        hb[ll * SH + cg * 2 + lh] = (f16)p[(size_t)cg * 2 * DSTRIDE];
    }
    __syncthreads();

    // y in f32 register D-tiles (lane = col, vi = row)
#pragma unroll
    for (int nt = 0; nt < 2; nt++)
#pragma unroll
      for (int vi = 0; vi < 8; vi++)
        yacc[nt][vi] = (float)hb[(vi + 8 * lh) * SH + nt * 16 + ll];

#pragma unroll 1
    for (int blk = 0; blk < 2; blk++) {
      if (blk == 1) {  // rebuild h = [y, coords]
#pragma unroll
        for (int nt = 0; nt < 2; nt++)
#pragma unroll
          for (int vi = 0; vi < 8; vi++)
            hb[(vi + 8 * lh) * SH + nt * 16 + ll] = (f16)yacc[nt][vi];
      }
      {  // copy coord features into cols 32..63 (16 halves per lane)
        const v8h* src = (const v8h*)(cb + ll * 32 + lh * 16);
        v8h c0 = src[0], c1 = src[1];
        v8h* dst = (v8h*)(hb + ll * SH + 32 + lh * 16);
        dst[0] = c0; dst[1] = c1;
      }
      __syncthreads();

#pragma unroll 1
      for (int l = 0; l < 3; l++) {
        v16h a0 = loadA(hb, 0, lh, ll);
        v16h a1 = loadA(hb, 1, lh, ll);
        const int mat = blk * 3 + l;
        const f16* wmat = wf + (size_t)mat * 4096 + lane * 16;
#pragma unroll
        for (int nt = 0; nt < 4; nt++) {
          int col = nt * 16 + ll;
          float bias = (col < HID) ? mlpB[mat * HID + col] : 0.f;
          v8f acc;
#pragma unroll
          for (int vi = 0; vi < 8; vi++) acc[vi] = bias;
          v16h b0 = *(const v16h*)(wmat + (size_t)(nt * 512));
          acc = wmma16(a0, b0, acc);
          v16h b1 = *(const v16h*)(wmat + (size_t)((4 + nt) * 512));
          acc = wmma16(a1, b1, acc);
#pragma unroll
          for (int vi = 0; vi < 8; vi++) acc[vi] = silu(acc[vi]);
#pragma unroll
          for (int vi = 0; vi < 8; vi++)
            hb[(vi + 8 * lh) * SH + nt * 16 + ll] = (f16)acc[vi];
        }
        __syncthreads();
      }
      {  // residual projection: y = y + h @ res_W + res_b
        v16h a0 = loadA(hb, 0, lh, ll);
        v16h a1 = loadA(hb, 1, lh, ll);
        const f16* wres = wf + RES_OFF + (size_t)blk * 2048 + lane * 16;
#pragma unroll
        for (int nt = 0; nt < 2; nt++) {
          v8f acc = yacc[nt];
          float bias = (blk == 0) ? bias_res[0][nt] : bias_res[1][nt];
#pragma unroll
          for (int vi = 0; vi < 8; vi++) acc[vi] += bias;
          v16h b0 = *(const v16h*)(wres + (size_t)(nt * 512));
          acc = wmma16(a0, b0, acc);
          v16h b1 = *(const v16h*)(wres + (size_t)((2 + nt) * 512));
          acc = wmma16(a1, b1, acc);
          yacc[nt] = acc;
        }
      }
      __syncthreads();
    }

    // post projection: (16x32) @ (32x48), weighted accumulate
#pragma unroll
    for (int nt = 0; nt < 2; nt++)
#pragma unroll
      for (int vi = 0; vi < 8; vi++)
        hb[(vi + 8 * lh) * SH + nt * 16 + ll] = (f16)yacc[nt][vi];
    __syncthreads();
    {
      v16h a = loadA(hb, 0, lh, ll);
      float wrow[8];
#pragma unroll
      for (int vi = 0; vi < 8; vi++) wrow[vi] = wb[vi + 8 * lh];
#pragma unroll
      for (int nt = 0; nt < 3; nt++) {
        v8f acc;
#pragma unroll
        for (int vi = 0; vi < 8; vi++) acc[vi] = bias_post[nt];
        v16h bm = *(const v16h*)(wf + POST_OFF + (size_t)(nt * 512) + lane * 16);
        acc = wmma16(a, bm, acc);
#pragma unroll
        for (int vi = 0; vi < 8; vi++) outacc[nt][vi] += wrow[vi] * acc[vi];
      }
    }
    __syncthreads();
  }

  // final store: out[b, col, qi]; each lane's 8 rows are contiguous in qi
#pragma unroll
  for (int nt = 0; nt < 3; nt++) {
    int col = nt * 16 + ll;
    if (col < OUTC) {
      float* op = out + ((size_t)b * OUTC + col) * NQ + qbase + 8 * lh;
      v4f lo, hi;
#pragma unroll
      for (int vi = 0; vi < 4; vi++) { lo[vi] = outacc[nt][vi]; hi[vi] = outacc[nt][vi + 4]; }
      *(v4f*)(op)     = lo;   // 16B aligned: qbase%16==0
      *(v4f*)(op + 4) = hi;
    }
  }
}

extern "C" void kernel_launch(void* const* d_in, const int* in_sizes, int n_in,
                              void* d_out, int out_size, void* d_ws, size_t ws_size,
                              hipStream_t stream) {
  const float* ctx   = (const float*)d_in[0];
  const float* query = (const float*)d_in[2];
  const float* Amat  = (const float*)d_in[3];
  const float* voxsz = (const float*)d_in[5];
  const float* mlpW  = (const float*)d_in[7];
  const float* mlpB  = (const float*)d_in[8];
  const float* resW  = (const float*)d_in[9];
  const float* resB  = (const float*)d_in[10];
  const float* postW = (const float*)d_in[11];
  const float* postB = (const float*)d_in[12];
  f16* wf = (f16*)d_ws;
  float* out = (float*)d_out;

  prep_weights<<<(WF_TOTAL + 255) / 256, 256, 0, stream>>>(mlpW, resW, postW, wf);

  int total_tiles = (2 * NQ) / 16;      // 32768 wave-tiles
  int blocks = total_tiles / WAVES;     // 4096 blocks x 256 threads (8 waves)
  decoder_main<<<blocks, 256, 0, stream>>>(ctx, query, Amat, voxsz,
                                           mlpB, resB, postB, wf, out);
}
